// PointGPTEncoder_21947282882908
// MI455X (gfx1250) — matrix operand order repaired and checked by reading, not compiled, not run on hardware
//
#include <hip/hip_runtime.h>
#include <hip/hip_bf16.h>
#include <math.h>

typedef __attribute__((ext_vector_type(16))) _Float16 v16h;
typedef __attribute__((ext_vector_type(8)))  float    v8f;

#define B_  32
#define N_  16384
#define G_  128
#define K_  64
#define D_  384
#define H_  128
#define LN_EPS_ 1e-5f

#define NTILES_ (D_/16)      // 24
#define KSTEPS_ (H_/32)      // 4
#define MTILES_ (K_/16)      // 4
#define TILES_  (MTILES_*NTILES_)   // 96

__device__ __forceinline__ float gelu_exact(float x) {
    return 0.5f * x * (1.0f + erff(x * 0.70710678118654752440f));
}

// ---------------------------------------------------------------------------
// Prep: w2 [128,384] fp32 -> f16, pre-swizzled into WMMA B-fragment layout:
//   [nt 0..23][kk 0..3][lane 0..31][16 halves], 32 bytes per lane per fragment.
// Lane l (half = l>=16, lr = l&15): column n = nt*16+lr; dword j packs
// K rows kbase + (half?16:0) + 2j and +1.
// ---------------------------------------------------------------------------
__global__ void k_w2_swizzle(const float* __restrict__ w2, _Float16* __restrict__ w2s) {
    const int tid = blockIdx.x * blockDim.x + threadIdx.x;
    if (tid >= NTILES_*KSTEPS_*32*8) return;
    const int j    = tid & 7;
    const int lane = (tid >> 3) & 31;
    const int f    = tid >> 8;           // fragment id = nt*4 + kk
    const int kk   = f & 3;
    const int nt   = f >> 2;
    const int half = (lane >= 16);
    const int lr   = lane & 15;
    const int n    = nt*16 + lr;
    const int k0   = kk*32 + (half ? 16 : 0) + 2*j;
    w2s[2*tid + 0] = (_Float16)w2[(size_t)k0     * D_ + n];
    w2s[2*tid + 1] = (_Float16)w2[(size_t)(k0+1) * D_ + n];
}

// ---------------------------------------------------------------------------
// Farthest point sampling: one block per batch, 128 sequential steps.
// Writes centroids [B,G,3] directly into d_out.
// ---------------------------------------------------------------------------
__global__ __launch_bounds__(256)
void k_fps(const float* __restrict__ points, float* __restrict__ dist,
           float* __restrict__ centroids_out) {
    const int b = blockIdx.x;
    const int tid = threadIdx.x;
    const float* P = points + (size_t)b * N_ * 3;
    float* dst = dist + (size_t)b * N_;

    __shared__ float s_val[256];
    __shared__ int   s_idx[256];
    __shared__ float s_c[3];
    __shared__ int   s_far;

    if (tid == 0) s_far = 0;
    for (int i = tid; i < N_; i += 256) dst[i] = 3.4e38f;
    __syncthreads();

    for (int s = 0; s < G_; ++s) {
        if (tid == 0) {
            int f = s_far;
            float cx = P[f*3+0], cy = P[f*3+1], cz = P[f*3+2];
            s_c[0] = cx; s_c[1] = cy; s_c[2] = cz;
            float* co = centroids_out + ((size_t)b * G_ + s) * 3;
            co[0] = cx; co[1] = cy; co[2] = cz;
        }
        __syncthreads();
        const float cx = s_c[0], cy = s_c[1], cz = s_c[2];
        float bv = -1.0f; int bi = 0x7fffffff;
        for (int i = tid; i < N_; i += 256) {
            float dx = P[i*3+0]-cx, dy = P[i*3+1]-cy, dz = P[i*3+2]-cz;
            float d  = dx*dx + dy*dy + dz*dz;
            float nd = fminf(dst[i], d);
            dst[i] = nd;
            if (nd > bv) { bv = nd; bi = i; }
        }
        s_val[tid] = bv; s_idx[tid] = bi;
        __syncthreads();
        for (int off = 128; off > 0; off >>= 1) {
            if (tid < off) {
                float ov = s_val[tid+off]; int oi = s_idx[tid+off];
                if (ov > s_val[tid] || (ov == s_val[tid] && oi < s_idx[tid])) {
                    s_val[tid] = ov; s_idx[tid] = oi;
                }
            }
            __syncthreads();
        }
        if (tid == 0) s_far = s_idx[0];
        __syncthreads();
    }
}

// ---------------------------------------------------------------------------
// kNN (64 smallest dist2) + local coords. One block per (b,g).
// dist2 computed as c2 + p2 - 2*c.p to match the reference's matmul form.
// ---------------------------------------------------------------------------
__global__ __launch_bounds__(256)
void k_group(const float* __restrict__ points,
             const float* __restrict__ centroids,   // read back from d_out
             float* __restrict__ lc) {              // [B*G, K, 3]
    const int bg  = blockIdx.x;
    const int b   = bg / G_;
    const int tid = threadIdx.x;
    const float* P = points + (size_t)b * N_ * 3;
    const float* C = centroids + (size_t)bg * 3;
    const float cx = C[0], cy = C[1], cz = C[2];
    const float c2 = cx*cx + cy*cy + cz*cz;

    float dv[64];
    const int base = tid * 64;
    #pragma unroll 4
    for (int j = 0; j < 64; ++j) {
        const int i = base + j;
        float px = P[i*3+0], py = P[i*3+1], pz = P[i*3+2];
        float p2 = px*px + py*py + pz*pz;
        dv[j] = c2 + p2 - 2.0f*(cx*px + cy*py + cz*pz);
    }
    unsigned long long used = 0ull;

    __shared__ float rv[256];
    __shared__ int   ri[256];
    __shared__ int   sel[K_];

    for (int r = 0; r < K_; ++r) {
        float bv = 3.4e38f; int bj = -1;
        for (int j = 0; j < 64; ++j) {
            if (!((used >> j) & 1ull) && dv[j] < bv) { bv = dv[j]; bj = j; }
        }
        rv[tid] = bv;
        ri[tid] = (bj < 0) ? 0x7fffffff : (base + bj);
        __syncthreads();
        for (int off = 128; off > 0; off >>= 1) {
            if (tid < off) {
                float ov = rv[tid+off]; int oi = ri[tid+off];
                if (ov < rv[tid] || (ov == rv[tid] && oi < ri[tid])) {
                    rv[tid] = ov; ri[tid] = oi;
                }
            }
            __syncthreads();
        }
        const int widx = ri[0];
        if (tid == 0) sel[r] = widx;
        if (widx >= base && widx < base + 64) used |= (1ull << (widx - base));
        __syncthreads();
    }

    if (tid < K_ * 3) {
        const int k = tid / 3, c = tid % 3;
        const int pi = sel[k];
        lc[((size_t)bg * K_ + k) * 3 + c] = P[pi*3 + c] - C[c];
    }
}

// ---------------------------------------------------------------------------
// MLP + mean + LayerNorm. One block (8 wave32) per (b,g).
// Layer1 (K-dim=3): scalar FMA + exact GELU -> f16, written into LDS
//   pre-swizzled as WMMA A fragments: [mt][kk][lane][16 halves].
// Layer2 (64x384x128): v_wmma_f32_16x16x32_f16; A = 2x ds_load_b128,
//   B = 2x global_load_b128 from the pre-swizzled w2s.
// Tile loop is scalar-controlled (readfirstlane wave id) so EXEC is all-1s
// at every WMMA by construction, as the ISA requires.
// Column sums -> mean over K=64 -> +b2 -> LayerNorm -> d_out.
// ---------------------------------------------------------------------------
__global__ __launch_bounds__(256)
void k_mlp(const float* __restrict__ lc,
           const float* __restrict__ w1, const float* __restrict__ b1,
           const _Float16* __restrict__ w2s, const float* __restrict__ b2,
           const float* __restrict__ lns, const float* __restrict__ lnb,
           float* __restrict__ tokens_out) {
    const int bg   = blockIdx.x;
    const int tid  = threadIdx.x;
    const int lane = tid & 31;
    const int wave = __builtin_amdgcn_readfirstlane(tid >> 5);   // SGPR

    // A fragments, swizzled: [mt 0..3][kk 0..3][lane 0..31][16 halves] = 16 KB
    __shared__ alignas(32) _Float16 shs[MTILES_*KSTEPS_*32*16];
    __shared__ float s_w1[3*H_];
    __shared__ float s_b1[H_];
    __shared__ float colsum[D_];
    __shared__ float tok[D_];
    __shared__ float rs[256];

    for (int i = tid; i < 3*H_; i += 256) s_w1[i] = w1[i];
    for (int i = tid; i < H_;   i += 256) s_b1[i] = b1[i];
    for (int i = tid; i < D_;   i += 256) colsum[i] = 0.0f;
    __syncthreads();

    // layer 1 + GELU (exact), scatter into A-fragment layout
    const float* LC = lc + (size_t)bg * K_ * 3;
    for (int e = tid; e < K_*H_; e += 256) {
        const int m = e >> 7;          // row 0..63
        const int c = e & (H_-1);      // hidden 0..127
        const float x0 = LC[m*3+0], x1 = LC[m*3+1], x2 = LC[m*3+2];
        float v = x0*s_w1[c] + x1*s_w1[H_+c] + x2*s_w1[2*H_+c] + s_b1[c];
        // swizzle (m,c) -> fragment (mt,kk), lane, half-slot q
        const int mt = m >> 4, lr = m & 15;
        const int kk = c >> 5, kw = c & 31;
        const int region = kw >> 3;                    // 0..3
        const int hsel   = kw >> 4;
        const int q      = kw & 15;
        const int ln     = hsel*16 + lr;
        shs[(((mt*KSTEPS_ + kk)*32) + ln)*16 + q] = (_Float16)gelu_exact(v);
    }
    __syncthreads();

    // layer 2: 4 M-tiles x 24 N-tiles, Kdim=128 in 4 steps of 32.
    // 12 tiles per wave; loop bounds are scalar -> no EXEC masking.
    const v16h* As = (const v16h*)shs;
    const v16h* Bs = (const v16h*)w2s;
    for (int t = wave; t < TILES_; t += 8) {
        const int mt = t & 3;
        const int nt = t >> 2;
        // prefetch next tile's B fragment (wraps; always in-bounds)
        {
            const int tn  = (t + 8 < TILES_) ? (t + 8) : wave;
            const int ntn = tn >> 2;
            __builtin_prefetch((const void*)(Bs + (ntn*KSTEPS_)*32 + lane), 0, 1);
        }
        v8f acc = {};
        #pragma unroll
        for (int kk = 0; kk < KSTEPS_; ++kk) {
            v16h a  = As[(mt*KSTEPS_ + kk)*32 + lane];   // 2x ds_load_b128
            v16h bf = Bs[(nt*KSTEPS_ + kk)*32 + lane];   // 2x global_load_b128
            acc = __builtin_amdgcn_wmma_f32_16x16x32_f16(
                      false, a, false, bf, (short)0, acc, false, false);
        }
        // per-lane sum over the 8 rows this lane holds -> column partial sum
        float ps = acc[0]+acc[1]+acc[2]+acc[3]+acc[4]+acc[5]+acc[6]+acc[7];
        atomicAdd(&colsum[nt*16 + (lane & 15)], ps);     // ds_add_f32
    }
    __syncthreads();

    // tokens = mean_k(feats) + b2
    for (int i = tid; i < D_; i += 256) tok[i] = colsum[i] * (1.0f/K_) + b2[i];
    __syncthreads();

    // LayerNorm (two-pass, matches reference mean/var form)
    float part = 0.0f;
    for (int i = tid; i < D_; i += 256) part += tok[i];
    rs[tid] = part; __syncthreads();
    for (int off = 128; off > 0; off >>= 1) {
        if (tid < off) rs[tid] += rs[tid+off];
        __syncthreads();
    }
    const float mu = rs[0] * (1.0f/D_);
    __syncthreads();
    part = 0.0f;
    for (int i = tid; i < D_; i += 256) { float d = tok[i]-mu; part += d*d; }
    rs[tid] = part; __syncthreads();
    for (int off = 128; off > 0; off >>= 1) {
        if (tid < off) rs[tid] += rs[tid+off];
        __syncthreads();
    }
    const float inv = rsqrtf(rs[0] * (1.0f/D_) + LN_EPS_);
    float* out = tokens_out + (size_t)bg * D_;
    for (int i = tid; i < D_; i += 256)
        out[i] = (tok[i]-mu) * inv * lns[i] + lnb[i];
}

// ---------------------------------------------------------------------------
extern "C" void kernel_launch(void* const* d_in, const int* in_sizes, int n_in,
                              void* d_out, int out_size, void* d_ws, size_t ws_size,
                              hipStream_t stream) {
    const float* points = (const float*)d_in[0];
    const float* w1     = (const float*)d_in[1];
    const float* b1     = (const float*)d_in[2];
    const float* w2     = (const float*)d_in[3];
    const float* b2     = (const float*)d_in[4];
    const float* lns    = (const float*)d_in[5];
    const float* lnb    = (const float*)d_in[6];

    float* out_centroids = (float*)d_out;                        // [B,G,3]
    float* out_tokens    = (float*)d_out + (size_t)B_*G_*3;      // [B,G,D]

    char* ws = (char*)d_ws;
    size_t off = 0;
    _Float16* w2s = (_Float16*)(ws + off);        // swizzled, 128*384*2 = 96 KB
    off += (size_t)H_ * D_ * sizeof(_Float16);
    off = (off + 255) & ~(size_t)255;
    float* dist = (float*)(ws + off);             // [B,N] = 2 MB
    off += (size_t)B_ * N_ * sizeof(float);
    off = (off + 255) & ~(size_t)255;
    float* lc = (float*)(ws + off);               // [B*G,K,3] = 3 MB

    k_w2_swizzle<<<(NTILES_*KSTEPS_*32*8 + 255)/256, 256, 0, stream>>>(w2, w2s);
    k_fps       <<<B_,     256, 0, stream>>>(points, dist, out_centroids);
    k_group     <<<B_*G_,  256, 0, stream>>>(points, out_centroids, lc);
    k_mlp       <<<B_*G_,  256, 0, stream>>>(lc, w1, b1, w2s, b2, lns, lnb, out_tokens);
}
